// VQVAE_31525059952911
// MI455X (gfx1250) — matrix operand order repaired and checked by reading, not compiled, not run on hardware
//
#include <hip/hip_runtime.h>
#include <hip/hip_bf16.h>

typedef __attribute__((ext_vector_type(16))) _Float16 v16h;
typedef __attribute__((ext_vector_type(8)))  _Float16 v8h;
typedef __attribute__((ext_vector_type(8)))  float    v8f;
typedef __attribute__((ext_vector_type(4)))  unsigned u32x4;
typedef __attribute__((ext_vector_type(8)))  unsigned u32x8;

#define WMMA_F16(a, b, c) \
    __builtin_amdgcn_wmma_f32_16x16x32_f16(false, (a), false, (b), (short)0, (c), false, false)

// ---------------------------------------------------------------------------
// Tensor Data Mover: 1-D contiguous tile load Global -> LDS.
// D# layout per CDNA5 ISA (group0 = 4 SGPRs, group1 = 8).
// All inputs must be wave-uniform (built from blockIdx + kernel args -> SALU).
// dsz_code: 0=1B, 1=2B, 2=4B. elems must be < 65536 (tile_dim0 is 16 bits).
// ---------------------------------------------------------------------------
__device__ __forceinline__ void tdm_load_1d(unsigned lds_off, const void* gptr,
                                            unsigned elems, unsigned dsz_code) {
    unsigned long long ga = (unsigned long long)gptr;
    u32x4 g0;
    g0[0] = 1u;                                            // count=1 (valid user D#)
    g0[1] = lds_off;                                       // LDS byte address
    g0[2] = (unsigned)ga;                                  // global_addr[31:0]
    g0[3] = (unsigned)((ga >> 32) & 0x01FFFFFFu) | (2u << 30); // addr[56:32] | type=2
    u32x8 g1;
    g1[0] = dsz_code << 16;            // workgroup_mask=0 | data_size
    g1[1] = (elems & 0xFFFFu) << 16;   // tensor_dim0[15:0]
    g1[2] = (elems >> 16) | (1u << 16);// tensor_dim0[31:16] | tensor_dim1 = 1
    g1[3] = (elems & 0xFFFFu) << 16;   // tensor_dim1 hi = 0 | tile_dim0 = elems
    g1[4] = 1u;                        // tile_dim1 = 1 | tile_dim2 = 0
    g1[5] = elems;                     // tensor_dim0_stride[31:0]
    g1[6] = 0u;                        // stride hi | tensor_dim1_stride lo
    g1[7] = 0u;
    asm volatile("tensor_load_to_lds %0, %1" :: "s"(g0), "s"(g1) : "memory");
}

// ---------------------------------------------------------------------------
// Prep kernels
// ---------------------------------------------------------------------------
__global__ void f32_to_f16_kernel(const float* __restrict__ in,
                                  _Float16* __restrict__ out, long n) {
    long t = (long)blockIdx.x * blockDim.x + threadIdx.x;
    if (t < n) out[t] = (_Float16)in[t];
}

// enc_w1 [128][3][4][4] -> [128][64] K-padded (k = ic*16 + kh*4 + kw, ic==3 -> 0)
__global__ void prep_w1_kernel(const float* __restrict__ w,
                               _Float16* __restrict__ wp) {
    int t = blockIdx.x * blockDim.x + threadIdx.x;
    if (t >= 128 * 64) return;
    int oc = t >> 6, k = t & 63;
    wp[t] = (k < 48) ? (_Float16)w[oc * 48 + k] : (_Float16)0.0f;
}

// cnorm[k] = 0.5 * || f16-rounded code ||^2  (match the f16 dot's operands)
__global__ void cnorm_kernel(const float* __restrict__ cb, float* __restrict__ cn) {
    int k = blockIdx.x * blockDim.x + threadIdx.x;
    if (k >= 512) return;
    float s = 0.f;
    for (int e = 0; e < 64; ++e) {
        float v = (float)(_Float16)cb[k * 64 + e];
        s += v * v;
    }
    cn[k] = 0.5f * s;
}

// deconv weights OIHW [OCsrc][IC][4][4] -> AW[cls=ph*2+pw][OCp][IC*4 + 2*i + j]
// with kh = ph + 2i, kw = pw + 2j; rows oc >= OCsrc zero-padded.
__global__ void prep_dw_kernel(const float* __restrict__ w,
                               _Float16* __restrict__ aw,
                               int OCsrc, int OCp, int IC) {
    int t = blockIdx.x * blockDim.x + threadIdx.x;
    int total = 4 * OCp * IC * 4;
    if (t >= total) return;
    int it  = t & 3;
    int ic  = (t >> 2) % IC;
    int oc  = ((t >> 2) / IC) % OCp;
    int cls = t / (OCp * IC * 4);
    int ph = cls >> 1, pw = cls & 1;
    int i = it >> 1, j = it & 1;
    _Float16 v = (_Float16)0.0f;
    if (oc < OCsrc)
        v = (_Float16)w[((long)oc * IC + ic) * 16 + (ph + 2 * i) * 4 + (pw + 2 * j)];
    aw[t] = v;
}

// ---------------------------------------------------------------------------
// Implicit-GEMM conv (k=4, s=2, p=1), WMMA f32 <- f16*f16.
// One block owns a 16-row M (oc) tile: wave0 TDM-stages the 16 x Kpad weight
// tile into LDS; the 8 waves each produce a 16(oc) x 16(pixel) output tile.
// K = ic*16 + kh*4 + kw (Kpad multiple of 32).
// ---------------------------------------------------------------------------
__global__ __launch_bounds__(256)
void conv_wmma_kernel(const _Float16* __restrict__ IN,   // [B][Cin][Hi][Wi]
                      const _Float16* __restrict__ W,    // [OC][Kpad]
                      const float* __restrict__ bias,    // [OC]
                      _Float16* __restrict__ OUT,        // [B][OC][Ho][Wo]
                      int B, int Cin, int Hi, int Wi,
                      int OC, int Ho, int Wo, int Kpad, int relu) {
    extern __shared__ _Float16 smem_c[];                  // [16][Kpad]
    const int lane = threadIdx.x & 31;
    const int wv   = threadIdx.x >> 5;
    const int hs   = lane >> 4;          // half-wave select
    const int l16  = lane & 15;
    const int nM   = OC >> 4;
    const int mt   = blockIdx.x % nM;
    const int ntb  = blockIdx.x / nM;
    const int oc0  = mt << 4;

    if (wv == 0) {
        tdm_load_1d(0u, W + (long)oc0 * Kpad, (unsigned)(16 * Kpad), 1u);
        __builtin_amdgcn_s_wait_tensorcnt(0);
    }
    __syncthreads();

    const int nt  = ntb * 8 + wv;
    const int n0  = nt << 4;
    const int HW  = Ho * Wo;
    const int b   = n0 / HW;
    const int rm  = n0 - b * HW;
    const int oh  = rm / Wo;
    const int ow  = (rm - oh * Wo) + l16;

    const int ihb = 2 * oh - 1;
    const int iwb = 2 * ow - 1;

    v8f acc = {};
    const _Float16* arow = smem_c + l16 * Kpad + hs * 16;

    for (int k0 = 0; k0 < Kpad; k0 += 32) {
        // A fragment from LDS: lane<16 -> K=k0+0..15; lane>=16 -> K=k0+16..31
        v8h alo = *(const v8h*)(arow + k0);
        v8h ahi = *(const v8h*)(arow + k0 + 8);
        v16h a;
#pragma unroll
        for (int i = 0; i < 8; ++i) { a[i] = alo[i]; a[8 + i] = ahi[i]; }

        // B fragment: one input channel's 4x4 patch for this lane's pixel.
        const int ic = (k0 >> 4) + hs;
        v16h bf;
        if (ic < Cin) {
            const _Float16* base = IN + (long)(b * Cin + ic) * Hi * Wi;
#pragma unroll
            for (int kh = 0; kh < 4; ++kh) {
                const int ih = ihb + kh;
                const bool okh = (ih >= 0) && (ih < Hi);
                const _Float16* rowp = base + (long)ih * Wi + iwb;
#pragma unroll
                for (int kw = 0; kw < 4; ++kw) {
                    const int iw = iwb + kw;
                    _Float16 v = (_Float16)0.0f;
                    if (okh && iw >= 0 && iw < Wi) v = rowp[kw];
                    bf[kh * 4 + kw] = v;
                }
            }
        } else {
#pragma unroll
            for (int t = 0; t < 16; ++t) bf[t] = (_Float16)0.0f;
        }
        acc = WMMA_F16(a, bf, acc);
    }

#pragma unroll
    for (int r = 0; r < 8; ++r) {
        const int M = r + (hs << 3);
        float v = acc[r] + bias[oc0 + M];
        if (relu) v = v > 0.f ? v : 0.f;
        OUT[(((long)b * OC + (oc0 + M)) * Ho + oh) * Wo + ow] = (_Float16)v;
    }
}

// ---------------------------------------------------------------------------
// Implicit-GEMM transposed conv (k=4, s=2, p=1): per output-parity class only
// a 2x2 set of taps is live -> K = IC*4 with per-class rearranged weights.
// Block owns (cls, mt); wave0 TDM-stages the 16 x Kdim weight tile into LDS.
// ---------------------------------------------------------------------------
template <typename OutT>
__global__ __launch_bounds__(256)
void deconv_wmma_kernel(const _Float16* __restrict__ IN,  // [B][IC][Hi][Wi]
                        const _Float16* __restrict__ AW,  // [4][OCp][IC*4]
                        const float* __restrict__ bias,   // [OC]
                        OutT* __restrict__ OUT,           // [B][OC][Ho][Wo]
                        int B, int IC, int Hi, int Wi,
                        int OCp, int OC, int Ho, int Wo, int relu) {
    extern __shared__ _Float16 smem_d[];                  // [16][Kdim]
    const int lane = threadIdx.x & 31;
    const int wv   = threadIdx.x >> 5;
    const int hs   = lane >> 4;
    const int l16  = lane & 15;
    const int Kdim = IC * 4;
    const int nM   = OCp >> 4;
    const int Hh = Ho >> 1, Wh = Wo >> 1;
    const int nNb  = ((B * Hh * Wh) >> 4) >> 3;           // N tiles / 8 waves
    const int mt   = blockIdx.x % nM;
    int tmp = blockIdx.x / nM;
    const int ntb  = tmp % nNb;
    const int cls  = tmp / nNb;
    const int ph = cls >> 1, pw = cls & 1;
    const int oc0  = mt << 4;

    if (wv == 0) {
        tdm_load_1d(0u, AW + ((long)cls * OCp + oc0) * Kdim,
                    (unsigned)(16 * Kdim), 1u);
        __builtin_amdgcn_s_wait_tensorcnt(0);
    }
    __syncthreads();

    const int nt  = ntb * 8 + wv;
    const int n0  = nt << 4;
    const int HWh = Hh * Wh;
    const int b   = n0 / HWh;
    int r2 = n0 - b * HWh;
    const int y = r2 / Wh;
    const int x = (r2 - y * Wh) + l16;

    v8f acc = {};
    const _Float16* arow = smem_d + l16 * Kdim + hs * 16;
    const _Float16* inb  = IN + (long)b * IC * Hi * Wi;

    const int ih0 = y + ph - 1;  // rows ih0, ih0+1
    const int iw0 = x + pw - 1;  // cols iw0, iw0+1

    for (int k0 = 0; k0 < Kdim; k0 += 32) {
        v8h alo = *(const v8h*)(arow + k0);
        v8h ahi = *(const v8h*)(arow + k0 + 8);
        v16h a;
#pragma unroll
        for (int i = 0; i < 8; ++i) { a[i] = alo[i]; a[8 + i] = ahi[i]; }

        // B fragment: 4 input channels x their 2x2 live patches.
        const int icb = (k0 >> 2) + (hs << 2);
        v16h bf;
#pragma unroll
        for (int u = 0; u < 4; ++u) {
            const _Float16* cbp = inb + (long)(icb + u) * Hi * Wi;
#pragma unroll
            for (int i = 0; i < 2; ++i) {
                const int ih = ih0 + i;
                const bool okh = (ih >= 0) && (ih < Hi);
#pragma unroll
                for (int j = 0; j < 2; ++j) {
                    const int iw = iw0 + j;
                    _Float16 v = (_Float16)0.0f;
                    if (okh && iw >= 0 && iw < Wi) v = cbp[(long)ih * Wi + iw];
                    bf[u * 4 + i * 2 + j] = v;
                }
            }
        }
        acc = WMMA_F16(a, bf, acc);
    }

    const int oh = ph + 2 * y;
    const int ow = pw + 2 * x;
#pragma unroll
    for (int r = 0; r < 8; ++r) {
        const int M  = r + (hs << 3);
        const int oc = oc0 + M;
        if (oc >= OC) continue;
        float v = acc[r] + bias[oc];
        if (relu) v = v > 0.f ? v : 0.f;
        OUT[(((long)b * OC + oc) * Ho + oh) * Wo + ow] = (OutT)v;
    }
}

// ---------------------------------------------------------------------------
// VQ argmin: GEMM codes(512) x vectors over K=64, score = 0.5||c||^2 - v.c
// Block TDM-stages the whole f16 codebook (64KB) + cnorm table into LDS;
// each of 8 waves handles 16 vectors, looping all 32 code tiles.
// ---------------------------------------------------------------------------
__global__ __launch_bounds__(256)
void vq_argmin_kernel(const _Float16* __restrict__ Z,   // [Nvec][64]
                      const _Float16* __restrict__ CB,  // [512][64]
                      const float* __restrict__ cnorm,  // [512]
                      int* __restrict__ idx, int Nvec) {
    extern __shared__ _Float16 smem_q[];  // [512*64] f16 codes + [512] f32 norms
    const int lane = threadIdx.x & 31;
    const int wv   = threadIdx.x >> 5;
    const int hs   = lane >> 4;
    const int l16  = lane & 15;

    if (wv == 0) {
        tdm_load_1d(0u, CB, 512u * 64u, 1u);          // f16 codes
        tdm_load_1d(512u * 64u * 2u, cnorm, 512u, 2u); // f32 0.5*||c||^2
        __builtin_amdgcn_s_wait_tensorcnt(0);
    }
    __syncthreads();
    const float* cnl = (const float*)(smem_q + 512 * 64);

    const int n0 = (blockIdx.x * 8 + wv) << 4;
    (void)Nvec;  // grid sized exactly

    const _Float16* zp = Z + (long)(n0 + l16) * 64;
    v16h b0 = *(const v16h*)(zp + hs * 16);       // K 0..31
    v16h b1 = *(const v16h*)(zp + 32 + hs * 16);  // K 32..63

    float best = 3.4e38f;
    int bidx = 0;
    for (int mt = 0; mt < 32; ++mt) {
        const int m0 = mt << 4;
        const _Float16* crow = smem_q + (m0 + l16) * 64 + hs * 16;
        v8h a0lo = *(const v8h*)(crow);
        v8h a0hi = *(const v8h*)(crow + 8);
        v8h a1lo = *(const v8h*)(crow + 32);
        v8h a1hi = *(const v8h*)(crow + 40);
        v16h a0, a1;
#pragma unroll
        for (int i = 0; i < 8; ++i) {
            a0[i] = a0lo[i]; a0[8 + i] = a0hi[i];
            a1[i] = a1lo[i]; a1[8 + i] = a1hi[i];
        }
        v8f acc = {};
        acc = WMMA_F16(a0, b0, acc);
        acc = WMMA_F16(a1, b1, acc);
#pragma unroll
        for (int r = 0; r < 8; ++r) {
            const int code = m0 + r + (hs << 3);
            const float s = cnl[code] - acc[r];
            if (s < best) { best = s; bidx = code; }  // ascending -> first min
        }
    }
    // combine the two half-wave partials for each vector (lanes j and j+16)
    float ob = __shfl_xor(best, 16, 32);
    int   oi = __shfl_xor(bidx, 16, 32);
    if (ob < best || (ob == best && oi < bidx)) { best = ob; bidx = oi; }
    if (lane < 16) idx[n0 + l16] = bidx;
}

// z_q[b][e][p] = CB[idx[b*4096 + p]][e],  p = h*64+w
__global__ void build_zq_kernel(const int* __restrict__ idx,
                                const _Float16* __restrict__ CB,
                                _Float16* __restrict__ ZQ, int B) {
    long t = (long)blockIdx.x * blockDim.x + threadIdx.x;
    long total = (long)B * 64 * 4096;
    if (t >= total) return;
    int p = (int)(t & 4095);
    int e = (int)((t >> 12) & 63);
    int b = (int)(t >> 18);
    int code = idx[b * 4096 + p];
    ZQ[t] = CB[code * 64 + e];
}

// ---------------------------------------------------------------------------
// Launch
// ---------------------------------------------------------------------------
extern "C" void kernel_launch(void* const* d_in, const int* in_sizes, int n_in,
                              void* d_out, int out_size, void* d_ws, size_t ws_size,
                              hipStream_t stream) {
    (void)in_sizes; (void)n_in; (void)out_size; (void)ws_size;
    const float* x      = (const float*)d_in[0];
    const float* enc_w1 = (const float*)d_in[1];
    const float* enc_b1 = (const float*)d_in[2];
    const float* enc_w2 = (const float*)d_in[3];
    const float* enc_b2 = (const float*)d_in[4];
    const float* dec_w1 = (const float*)d_in[5];
    const float* dec_b1 = (const float*)d_in[6];
    const float* dec_w2 = (const float*)d_in[7];
    const float* dec_b2 = (const float*)d_in[8];
    const float* cbook  = (const float*)d_in[9];
    float* out = (float*)d_out;

    const int B = 32;

    // workspace carve-up (256B aligned); d1 aliases h1 (dead by then)
    char* ws = (char*)d_ws;
    size_t off = 0;
    auto take = [&](size_t bytes) {
        char* p = ws + off;
        off += (bytes + 255) & ~(size_t)255;
        return p;
    };
    _Float16* x_h  = (_Float16*)take((size_t)B * 3 * 256 * 256 * 2);
    _Float16* h1_h = (_Float16*)take((size_t)B * 128 * 128 * 128 * 2);
    _Float16* z_h  = (_Float16*)take((size_t)B * 64 * 64 * 64 * 2);
    _Float16* zq_h = (_Float16*)take((size_t)B * 64 * 64 * 64 * 2);
    _Float16* w1p  = (_Float16*)take(128 * 64 * 2);
    _Float16* w2h  = (_Float16*)take(64 * 2048 * 2);
    _Float16* cbh  = (_Float16*)take(512 * 64 * 2);
    float*    cnm  = (float*)take(512 * 4);
    int*      idxb = (int*)take((size_t)B * 4096 * 4);
    _Float16* aw1  = (_Float16*)take(4 * 128 * 256 * 2);
    _Float16* aw2  = (_Float16*)take(4 * 16 * 512 * 2);
    _Float16* d1_h = h1_h;  // alias: h1 is dead once conv2 finished

    auto cdiv = [](long a, long b) { return (int)((a + b - 1) / b); };

    // --- prep ---
    long nx = (long)B * 3 * 256 * 256;
    f32_to_f16_kernel<<<cdiv(nx, 256), 256, 0, stream>>>(x, x_h, nx);
    prep_w1_kernel<<<cdiv(128 * 64, 256), 256, 0, stream>>>(enc_w1, w1p);
    f32_to_f16_kernel<<<cdiv(64 * 2048, 256), 256, 0, stream>>>(enc_w2, w2h, 64 * 2048);
    f32_to_f16_kernel<<<cdiv(512 * 64, 256), 256, 0, stream>>>(cbook, cbh, 512 * 64);
    cnorm_kernel<<<2, 256, 0, stream>>>(cbook, cnm);
    prep_dw_kernel<<<cdiv(4 * 128 * 64 * 4, 256), 256, 0, stream>>>(dec_w1, aw1, 128, 128, 64);
    prep_dw_kernel<<<cdiv(4 * 16 * 128 * 4, 256), 256, 0, stream>>>(dec_w2, aw2, 3, 16, 128);

    // --- encoder ---
    {   // conv1: 3 -> 128, 256 -> 128; blocks = nM * nN/8, LDS = 16*64*2 B
        int nM = 128 / 16;
        long nN = (long)B * 128 * 128 / 16;
        conv_wmma_kernel<<<(int)(nM * (nN / 8)), 256, 16 * 64 * 2, stream>>>(
            x_h, w1p, enc_b1, h1_h, B, 3, 256, 256, 128, 128, 128, 64, 1);
    }
    {   // conv2: 128 -> 64, 128 -> 64; LDS = 16*2048*2 = 64KB
        int nM = 64 / 16;
        long nN = (long)B * 64 * 64 / 16;
        conv_wmma_kernel<<<(int)(nM * (nN / 8)), 256, 16 * 2048 * 2, stream>>>(
            h1_h, w2h, enc_b2, z_h, B, 128, 128, 128, 64, 64, 64, 2048, 0);
    }

    // --- vector quantization ---
    {
        int Nvec = B * 4096;
        int blocks = Nvec / 128;  // 8 waves x 16 vectors per block
        size_t lds = 512 * 64 * 2 + 512 * 4;
        vq_argmin_kernel<<<blocks, 256, lds, stream>>>(z_h, cbh, cnm, idxb, Nvec);
        long nzq = (long)B * 64 * 4096;
        build_zq_kernel<<<cdiv(nzq, 256), 256, 0, stream>>>(idxb, cbh, zq_h, B);
    }

    // --- decoder ---
    {   // deconv1: 64 -> 128, 64 -> 128, ReLU, f16 out; LDS = 16*256*2 = 8KB
        int nM = 128 / 16;
        long nNb = ((long)B * 64 * 64 / 16) / 8;
        deconv_wmma_kernel<_Float16><<<(int)(4 * nM * nNb), 256, 16 * 256 * 2, stream>>>(
            zq_h, aw1, dec_b1, d1_h, B, 64, 64, 64, 128, 128, 128, 128, 1);
    }
    {   // deconv2: 128 -> 3 (padded to 16), 128 -> 256, f32 out; LDS = 16KB
        int nM = 16 / 16;
        long nNb = ((long)B * 128 * 128 / 16) / 8;
        deconv_wmma_kernel<float><<<(int)(4 * nM * nNb), 256, 16 * 512 * 2, stream>>>(
            d1_h, aw2, dec_b2, out, B, 128, 128, 128, 16, 3, 256, 256, 0);
    }
}
